// MixtralDecoderLayer_87462714016324
// MI455X (gfx1250) — matrix operand order repaired and checked by reading, not compiled, not run on hardware
//
#include <hip/hip_runtime.h>
#include <hip/hip_bf16.h>

#define H    2048
#define NH   16
#define NKV  8
#define HD   128
#define E    8
#define F    4096
#define T    2048
#define QKVW (NH*HD + 2*NKV*HD)   // 4096
#define EPSV 1e-5f

typedef __attribute__((ext_vector_type(16))) __bf16 v16bf;
typedef __attribute__((ext_vector_type(8)))  __bf16 v8bf;
typedef __attribute__((ext_vector_type(8)))  float  v8f;
typedef __attribute__((ext_vector_type(4)))  int    v4i;

typedef __attribute__((address_space(1))) v4i* gv4i_ptr;   // global int4*
typedef __attribute__((address_space(3))) v4i* lv4i_ptr;   // LDS int4*

// CDNA5 async global->LDS path (compile-safe probe)
#if defined(__has_builtin)
#  if __has_builtin(__builtin_amdgcn_global_load_async_to_lds_b128)
#    define HAVE_ASYNC_LDS 1
#  endif
#endif
#ifndef HAVE_ASYNC_LDS
#  define HAVE_ASYNC_LDS 0
#endif

__device__ __forceinline__ void wait_async_zero() {
#if defined(__has_builtin)
#  if __has_builtin(__builtin_amdgcn_s_wait_asynccnt)
    __builtin_amdgcn_s_wait_asynccnt(0);
    return;
#  else
    asm volatile("s_wait_asynccnt 0x0" ::: "memory");
    return;
#  endif
#else
    asm volatile("s_wait_asynccnt 0x0" ::: "memory");
#endif
}

__device__ __forceinline__ unsigned short f32_to_bf16(float f) {
    unsigned int u = __float_as_uint(f);
    u += 0x7fffu + ((u >> 16) & 1u);       // round-to-nearest-even
    return (unsigned short)(u >> 16);
}

// ---------------------------------------------------------------------------
// WMMA bf16 GEMM:  C[M,N] = A[M,K] (bf16) x B[K,N] (f32 -> bf16 on stage-in)
// N, K compile-time -> all strides become immediate offsets.
// Tiles: BM=128, BN=128, BK=64 (two WMMA k-steps per LDS stage).
// 256 threads = 8 waves in a 2x4 grid; each wave owns a 64x32 patch
// = 4x2 WMMA 16x16 tiles -> 16 v_wmma per stage, 2 barriers per stage.
// A tile staged via GLOBAL_LOAD_ASYNC_TO_LDS_B128 when available.
// ---------------------------------------------------------------------------
#define BM 128
#define BN 128
#define BK 64
#define LDP (BK + 8)   // padded row stride (144B): de-conflicts ds_load_b128

template<int N, int K>
__global__ __launch_bounds__(256)
void gemm_bf16_wmma(const unsigned short* __restrict__ A,
                    const float* __restrict__ B,
                    float* __restrict__ C) {
    __shared__ unsigned short As[BM][LDP];   // (m, k)  ~18 KB
    __shared__ unsigned short Bs[BN][LDP];   // (n, k)  ~18 KB

    const int tid  = threadIdx.x;
    const int lane = tid & 31;
    const int wid  = tid >> 5;
    const int wm   = wid >> 2;              // 0..1  (64 rows each)
    const int wn   = wid & 3;               // 0..3  (32 cols each)
    const int m0   = blockIdx.y * BM;
    const int n0   = blockIdx.x * BN;

    const v8f vzero = {0.f,0.f,0.f,0.f,0.f,0.f,0.f,0.f};
    v8f acc[4][2];
#pragma unroll
    for (int i = 0; i < 4; ++i)
#pragma unroll
        for (int j = 0; j < 2; ++j) acc[i][j] = vzero;

    // A staging: 128x64 ushorts = 1024 16B-chunks; 4 chunks/thread.
    // chunk c: row = c>>3, col = (c&7)*8  -> per-thread constant (row*K+col).
    const unsigned short* aBase = A + (size_t)m0 * K;
    // B staging: thread owns column nc, rows kr0..kr0+31 (stride N = imm).
    const int nc  = tid & 127;
    const int kr0 = (tid >> 7) * 32;
    const float* bBase = B + n0 + nc;

    const int ktiles = K / BK;
    for (int kt = 0; kt < ktiles; ++kt) {
        const int k0 = kt * BK;

        // ---- stage A (bf16 byte-copy: async DMA to LDS when available) ----
#pragma unroll
        for (int i = 0; i < 4; ++i) {
            const int c   = tid + i * 256;
            const int row = c >> 3;
            const int col = (c & 7) * 8;
#if HAVE_ASYNC_LDS
            __builtin_amdgcn_global_load_async_to_lds_b128(
                (gv4i_ptr)(aBase + (size_t)row * K + k0 + col),
                (lv4i_ptr)(&As[row][col]),
                0, 0);
#else
            *(uint4*)(&As[row][col]) =
                *(const uint4*)(aBase + (size_t)row * K + k0 + col);
#endif
        }

        // ---- stage B (f32 -> bf16, transposed; 32 imm-offset loads) ----
        {
            const float* bp = bBase + (size_t)k0 * N + (size_t)kr0 * N;
#pragma unroll
            for (int i = 0; i < 32; ++i)
                Bs[nc][kr0 + i] = f32_to_bf16(bp[(size_t)i * N]);
        }

        // prefetch next k-tile into near caches
        if (kt + 1 < ktiles) {
            __builtin_prefetch(aBase + (size_t)(tid >> 3) * K + k0 + BK, 0, 3);
            __builtin_prefetch(bBase + (size_t)(k0 + BK + kr0) * N, 0, 3);
        }

#if HAVE_ASYNC_LDS
        wait_async_zero();
#endif
        __syncthreads();

        // ---- 2 WMMA k-steps out of this stage ----
        const int mlane = lane & 15;
        const int abase = (lane < 16) ? 0 : 8;    // A: K 0..7/16..23 vs 8..15/24..31
        const int bbase = (lane < 16) ? 0 : 8;    // B: K 0..7/16..23 vs 8..15/24..31
#pragma unroll
        for (int ks = 0; ks < 2; ++ks) {
            const int ko = ks * 32;
            v16bf bfrag[2];
#pragma unroll
            for (int tn = 0; tn < 2; ++tn) {
                const unsigned short* p = &Bs[wn * 32 + tn * 16 + mlane][ko + bbase];
                v8bf lo = *(const v8bf*)p;
                v8bf hi = *(const v8bf*)(p + 16);
                bfrag[tn] = __builtin_shufflevector(lo, hi,
                    0,1,2,3,4,5,6,7,8,9,10,11,12,13,14,15);
            }
#pragma unroll
            for (int tm = 0; tm < 4; ++tm) {
                const unsigned short* p = &As[wm * 64 + tm * 16 + mlane][ko + abase];
                v8bf lo = *(const v8bf*)p;
                v8bf hi = *(const v8bf*)(p + 16);
                v16bf afrag = __builtin_shufflevector(lo, hi,
                    0,1,2,3,4,5,6,7,8,9,10,11,12,13,14,15);
#pragma unroll
                for (int tn = 0; tn < 2; ++tn)
                    acc[tm][tn] = __builtin_amdgcn_wmma_f32_16x16x32_bf16(
                        false, afrag, false, bfrag[tn],
                        (short)0, acc[tm][tn], false, false);
            }
        }
        __syncthreads();
    }

    // ---- store C: VGPR r holds M = r (lanes 0-15) / M = 8+r (lanes 16-31) ----
    const int rowsel = (lane >> 4) * 8;
    const int coll   = lane & 15;
#pragma unroll
    for (int tm = 0; tm < 4; ++tm)
#pragma unroll
        for (int tn = 0; tn < 2; ++tn) {
            float* cp = C + (size_t)(m0 + wm * 64 + tm * 16 + rowsel) * N
                          + (n0 + wn * 32 + tn * 16 + coll);
#pragma unroll
            for (int r = 0; r < 8; ++r)
                cp[(size_t)r * N] = acc[tm][tn][r];
        }
}

// ---------------------------------------------------------------------------
__global__ __launch_bounds__(256)
void rmsnorm_kernel(const float* __restrict__ x, const float* __restrict__ w,
                    float* __restrict__ yf, unsigned short* __restrict__ ybf) {
    const int t = blockIdx.x;
    const int tid = threadIdx.x;
    const float* xr = x + (size_t)t * H;
    __shared__ float red[256];
    float ss = 0.f;
    for (int i = tid; i < H; i += 256) { float v = xr[i]; ss += v * v; }
    red[tid] = ss; __syncthreads();
    for (int s = 128; s > 0; s >>= 1) {
        if (tid < s) red[tid] += red[tid + s];
        __syncthreads();
    }
    const float inv = rsqrtf(red[0] / (float)H + EPSV);
    for (int i = tid; i < H; i += 256) {
        float v = xr[i] * inv * w[i];
        yf[(size_t)t * H + i]  = v;
        ybf[(size_t)t * H + i] = f32_to_bf16(v);
    }
}

// ---------------------------------------------------------------------------
__global__ __launch_bounds__(256)
void rope_kernel(float* __restrict__ qkv, const int* __restrict__ positions) {
    const int t = blockIdx.x;
    const float pos = (float)positions[t];
    float* row = qkv + (size_t)t * QKVW;
    const float LOGB = 9.210340371976184f;          // ln(10000)
    for (int p = threadIdx.x; p < (NH + NKV) * (HD / 2); p += 256) {
        const int head = p >> 6;                     // 0..23
        const int i    = p & 63;
        const int off  = (head < NH) ? head * HD
                                     : NH * HD + (head - NH) * HD;
        const float inv_freq = __expf(-(float)i * (LOGB / 64.f));
        const float ang = pos * inv_freq;
        const float c = cosf(ang), s = sinf(ang);
        const float x1 = row[off + i], x2 = row[off + i + 64];
        row[off + i]      = x1 * c - x2 * s;
        row[off + i + 64] = x2 * c + x1 * s;
    }
}

// ---------------------------------------------------------------------------
// Per-(head, query) causal softmax attention; scores staged in LDS.
__global__ __launch_bounds__(256)
void attn_kernel(const float* __restrict__ qkv,
                 unsigned short* __restrict__ attn_bf) {
    const int h   = blockIdx.x;
    const int t   = blockIdx.y;
    const int kvh = h >> 1;                 // rep = NH/NKV = 2
    const int tid = threadIdx.x;

    __shared__ float qs[HD];
    __shared__ float sc[T];
    __shared__ float red[256];

    if (tid < HD) qs[tid] = qkv[(size_t)t * QKVW + h * HD + tid];
    __syncthreads();

    const float scale = 0.08838834764831845f;   // 128^-0.5
    float lmax = -3.4e38f;
    for (int s = tid; s <= t; s += 256) {
        const float* kp = qkv + (size_t)s * QKVW + NH * HD + kvh * HD;
        float d = 0.f;
#pragma unroll 8
        for (int i = 0; i < HD; ++i) d += qs[i] * kp[i];
        d *= scale;
        sc[s] = d;
        lmax = fmaxf(lmax, d);
    }
    red[tid] = lmax; __syncthreads();
    for (int s2 = 128; s2 > 0; s2 >>= 1) {
        if (tid < s2) red[tid] = fmaxf(red[tid], red[tid + s2]);
        __syncthreads();
    }
    const float m = red[0];
    __syncthreads();

    float lsum = 0.f;
    for (int s = tid; s <= t; s += 256) {
        float p = __expf(sc[s] - m);
        sc[s] = p;
        lsum += p;
    }
    red[tid] = lsum; __syncthreads();
    for (int s2 = 128; s2 > 0; s2 >>= 1) {
        if (tid < s2) red[tid] += red[tid + s2];
        __syncthreads();
    }
    const float invl = 1.f / red[0];

    if (tid < HD) {
        float a = 0.f;
        const size_t vbase = (size_t)(NH * HD + NKV * HD) + (size_t)kvh * HD + tid;
        for (int s = 0; s <= t; ++s)
            a += sc[s] * qkv[(size_t)s * QKVW + vbase];
        attn_bf[(size_t)t * (NH * HD) + h * HD + tid] = f32_to_bf16(a * invl);
    }
}

// ---------------------------------------------------------------------------
__global__ __launch_bounds__(256)
void gate_kernel(const float* __restrict__ hsn2, const float* __restrict__ gw,
                 float* __restrict__ dense_w) {
    const int t = blockIdx.x;
    const int tid = threadIdx.x;
    float acc[E] = {0.f,0.f,0.f,0.f,0.f,0.f,0.f,0.f};
    const float* xr = hsn2 + (size_t)t * H;
    for (int i = tid; i < H; i += 256) {
        float x = xr[i];
#pragma unroll
        for (int e = 0; e < E; ++e) acc[e] += x * gw[i * E + e];
    }
    __shared__ float red[256];
    __shared__ float logits[E];
    for (int e = 0; e < E; ++e) {
        red[tid] = acc[e]; __syncthreads();
        for (int s = 128; s > 0; s >>= 1) {
            if (tid < s) red[tid] += red[tid + s];
            __syncthreads();
        }
        if (tid == 0) logits[e] = red[0];
        __syncthreads();
    }
    if (tid == 0) {
        float mx = logits[0];
        for (int e = 1; e < E; ++e) mx = fmaxf(mx, logits[e]);
        float p[E], sum = 0.f;
        for (int e = 0; e < E; ++e) { p[e] = __expf(logits[e] - mx); sum += p[e]; }
        for (int e = 0; e < E; ++e) p[e] /= sum;
        int i0 = 0;
        for (int e = 1; e < E; ++e) if (p[e] > p[i0]) i0 = e;
        int i1 = -1;
        for (int e = 0; e < E; ++e) {
            if (e == i0) continue;
            if (i1 < 0 || p[e] > p[i1]) i1 = e;
        }
        const float s2 = p[i0] + p[i1];
        for (int e = 0; e < E; ++e) dense_w[(size_t)t * E + e] = 0.f;
        dense_w[(size_t)t * E + i0] = p[i0] / s2;
        dense_w[(size_t)t * E + i1] = p[i1] / s2;
    }
}

// ---------------------------------------------------------------------------
__global__ __launch_bounds__(256)
void add_residual_kernel(const float* __restrict__ a, const float* __restrict__ b,
                         float* __restrict__ o, int n) {
    for (int i = blockIdx.x * 256 + threadIdx.x; i < n; i += gridDim.x * 256)
        o[i] = a[i] + b[i];
}

__global__ __launch_bounds__(256)
void silu_mul_kernel(const float* __restrict__ h1, const float* __restrict__ h3,
                     unsigned short* __restrict__ hp, int n) {
    for (int i = blockIdx.x * 256 + threadIdx.x; i < n; i += gridDim.x * 256) {
        float v = h1[i];
        float s = v / (1.f + __expf(-v));
        hp[i] = f32_to_bf16(s * h3[i]);
    }
}

__global__ __launch_bounds__(256)
void scale_accum_kernel(float* __restrict__ out, const float* __restrict__ y,
                        const float* __restrict__ dw, int e, int n) {
    for (int i = blockIdx.x * 256 + threadIdx.x; i < n; i += gridDim.x * 256) {
        const int t = i / H;
        out[i] += y[i] * dw[(size_t)t * E + e];
    }
}

__global__ __launch_bounds__(256)
void zero_kernel(float* __restrict__ o, int n) {
    for (int i = blockIdx.x * 256 + threadIdx.x; i < n; i += gridDim.x * 256)
        o[i] = 0.f;
}

// ---------------------------------------------------------------------------
extern "C" void kernel_launch(void* const* d_in, const int* in_sizes, int n_in,
                              void* d_out, int out_size, void* d_ws, size_t ws_size,
                              hipStream_t stream) {
    (void)in_sizes; (void)n_in; (void)out_size; (void)ws_size;

    const int*   positions = (const int*)  d_in[0];
    const float* hidden    = (const float*)d_in[1];
    const float* ln1_w     = (const float*)d_in[2];
    const float* ln2_w     = (const float*)d_in[3];
    const float* wqkv      = (const float*)d_in[4];
    const float* wo        = (const float*)d_in[5];
    const float* gate_w    = (const float*)d_in[6];
    const float* w1        = (const float*)d_in[7];
    const float* w2        = (const float*)d_in[8];
    const float* w3        = (const float*)d_in[9];

    float* out_hs  = (float*)d_out;                       // (hs)
    float* out_res = (float*)d_out + (size_t)T * H;       // (residual)

    // workspace carve-up (256B aligned)
    char* base = (char*)d_ws;
    size_t off = 0;
    auto take = [&](size_t bytes) -> char* {
        char* r = base + off;
        off = (off + bytes + 255) & ~(size_t)255;
        return r;
    };
    const size_t TH = (size_t)T * H;
    float*          hsn1f   = (float*)         take(TH * 4);
    unsigned short* hsn1bf  = (unsigned short*)take(TH * 2);
    float*          qkv     = (float*)         take((size_t)T * QKVW * 4);
    unsigned short* attnbf  = (unsigned short*)take(TH * 2);
    float*          hsn2f   = (float*)         take(TH * 4);
    unsigned short* hsn2bf  = (unsigned short*)take(TH * 2);
    float*          dense_w = (float*)         take((size_t)T * E * 4);
    float*          h1buf   = (float*)         take((size_t)T * F * 4);
    float*          h3buf   = (float*)         take((size_t)T * F * 4);
    unsigned short* hpbf    = (unsigned short*)take((size_t)T * F * 2);
    float*          ybuf    = (float*)         take(TH * 4);

    const dim3 blk(256);
    const dim3 ew_grid(2048);

    // 1) ln1
    rmsnorm_kernel<<<T, blk, 0, stream>>>(hidden, ln1_w, hsn1f, hsn1bf);
    // 2) qkv = hsn1 @ wqkv   (2048 x 4096 x 2048)
    gemm_bf16_wmma<QKVW, H><<<dim3(QKVW / BN, T / BM), blk, 0, stream>>>(
        hsn1bf, wqkv, qkv);
    // 3) RoPE on q,k
    rope_kernel<<<T, blk, 0, stream>>>(qkv, positions);
    // 4) attention -> bf16
    attn_kernel<<<dim3(NH, T), blk, 0, stream>>>(qkv, attnbf);
    // 5) o-proj + residual
    gemm_bf16_wmma<H, NH * HD><<<dim3(H / BN, T / BM), blk, 0, stream>>>(
        attnbf, wo, ybuf);
    add_residual_kernel<<<ew_grid, blk, 0, stream>>>(ybuf, hidden, out_res, (int)TH);
    // 6) ln2 + gate
    rmsnorm_kernel<<<T, blk, 0, stream>>>(out_res, ln2_w, hsn2f, hsn2bf);
    gate_kernel<<<T, blk, 0, stream>>>(hsn2f, gate_w, dense_w);
    // 7) MoE (dense over experts, scaled by routing weights — matches reference)
    zero_kernel<<<ew_grid, blk, 0, stream>>>(out_hs, (int)TH);
    for (int e = 0; e < E; ++e) {
        const float* w1e = w1 + (size_t)e * H * F;
        const float* w3e = w3 + (size_t)e * H * F;
        const float* w2e = w2 + (size_t)e * F * H;
        gemm_bf16_wmma<F, H><<<dim3(F / BN, T / BM), blk, 0, stream>>>(
            hsn2bf, w1e, h1buf);
        gemm_bf16_wmma<F, H><<<dim3(F / BN, T / BM), blk, 0, stream>>>(
            hsn2bf, w3e, h3buf);
        silu_mul_kernel<<<ew_grid, blk, 0, stream>>>(h1buf, h3buf, hpbf, T * F);
        gemm_bf16_wmma<H, F><<<dim3(H / BN, T / BM), blk, 0, stream>>>(
            hpbf, w2e, ybuf);
        scale_accum_kernel<<<ew_grid, blk, 0, stream>>>(out_hs, ybuf, dense_w, e, (int)TH);
    }
}
